// ProposalModule_523986010523
// MI455X (gfx1250) — matrix operand order repaired and checked by reading, not compiled, not run on hardware
//
#include <hip/hip_runtime.h>

typedef unsigned short u16;
typedef unsigned int   u32;
typedef __attribute__((ext_vector_type(16))) __bf16 v16bf;
typedef __attribute__((ext_vector_type(8)))  float  v8f;

union FragU { u32 u[8]; v16bf v; };

static constexpr int cB = 8, cK = 8192, cC = 128, cP = 1024, cS = 16;
static constexpr int cOUT = 79;
static constexpr float cRAD = 0.3f;
static constexpr int KP0 = 160;   // 131 channels padded to 5*32
static constexpr int CH  = 128;

// ---- workspace layout (bytes) ----
static constexpr size_t OFF_IND  = 0;
static constexpr size_t OFF_NXYZ = OFF_IND  + (size_t)cB*cP*4;
static constexpr size_t OFF_IDX  = OFF_NXYZ + (size_t)cB*cP*3*4;
static constexpr size_t OFF_WB0  = OFF_IDX  + (size_t)cB*cP*cS*4;
static constexpr size_t OFF_BB0  = OFF_WB0 + (size_t)128*KP0*2;
static constexpr size_t OFF_WB1  = OFF_BB0 + 128*4;
static constexpr size_t OFF_BB1  = OFF_WB1 + (size_t)128*128*2;
static constexpr size_t OFF_WB2  = OFF_BB1 + 128*4;
static constexpr size_t OFF_BB2  = OFF_WB2 + (size_t)128*128*2;
static constexpr size_t OFF_WH1  = OFF_BB2 + 128*4;
static constexpr size_t OFF_BH1  = OFF_WH1 + (size_t)128*128*2;
static constexpr size_t OFF_WH2  = OFF_BH1 + 128*4;
static constexpr size_t OFF_BH2  = OFF_WH2 + (size_t)128*128*2;
static constexpr size_t OFF_WH3  = OFF_BH2 + 128*4;
static constexpr size_t OFF_BH3  = OFF_WH3 + (size_t)80*128*2;
static constexpr size_t OFF_FEAT = ((OFF_BH3 + 80*4) + 255) & ~(size_t)255;

__device__ __forceinline__ u16 f2bf(float f) {
  u32 x = __float_as_uint(f);
  return (u16)((x + 0x7fffu + ((x >> 16) & 1u)) >> 16);   // RNE
}

// ---------------------------------------------------------------------------
// Weight prep: fold per-channel scales into weight rows, convert to bf16.
// ---------------------------------------------------------------------------
__global__ __launch_bounds__(256) void prep_kernel(
    const float* __restrict__ w0, const float* __restrict__ w1, const float* __restrict__ w2,
    const float* __restrict__ g0, const float* __restrict__ b0,
    const float* __restrict__ g1, const float* __restrict__ b1,
    const float* __restrict__ g2, const float* __restrict__ b2,
    const float* __restrict__ wc1, const float* __restrict__ bc1,
    const float* __restrict__ gn1, const float* __restrict__ bn1,
    const float* __restrict__ wc2, const float* __restrict__ bc2,
    const float* __restrict__ gn2, const float* __restrict__ bn2,
    const float* __restrict__ wc3, const float* __restrict__ bc3,
    u16* __restrict__ Wb0, float* __restrict__ Bb0,
    u16* __restrict__ Wb1, float* __restrict__ Bb1,
    u16* __restrict__ Wb2, float* __restrict__ Bb2,
    u16* __restrict__ Wh1, float* __restrict__ Bh1,
    u16* __restrict__ Wh2, float* __restrict__ Bh2,
    u16* __restrict__ Wh3, float* __restrict__ Bh3) {
  const int gt = blockIdx.x * blockDim.x + threadIdx.x;
  const int st = gridDim.x * blockDim.x;
  for (int i = gt; i < 128 * KP0; i += st) {
    int o = i / KP0, c = i - o * KP0;
    Wb0[i] = f2bf(c < 131 ? g0[o] * w0[o * 131 + c] : 0.0f);
  }
  for (int i = gt; i < 128 * 128; i += st) {
    int o = i >> 7;
    Wb1[i] = f2bf(g1[o]  * w1[i]);
    Wb2[i] = f2bf(g2[o]  * w2[i]);
    Wh1[i] = f2bf(gn1[o] * wc1[i]);
    Wh2[i] = f2bf(gn2[o] * wc2[i]);
  }
  for (int i = gt; i < 80 * 128; i += st) {
    int o = i >> 7;
    Wh3[i] = f2bf(o < cOUT ? wc3[i] : 0.0f);
  }
  if (gt < 128) {
    Bb0[gt] = b0[gt];
    Bb1[gt] = b1[gt];
    Bb2[gt] = b2[gt];
    Bh1[gt] = gn1[gt] * bc1[gt] + bn1[gt];
    Bh2[gt] = gn2[gt] * bc2[gt] + bn2[gt];
  }
  if (gt < 80) Bh3[gt] = (gt < cOUT) ? bc3[gt] : 0.0f;
}

// ---------------------------------------------------------------------------
// Farthest point sampling: one block per batch, points live in registers.
// ---------------------------------------------------------------------------
__global__ __launch_bounds__(256) void fps_kernel(const float* __restrict__ xyz,
                                                  int* __restrict__ inds,
                                                  float* __restrict__ nxyz) {
  const int b = blockIdx.x, tid = threadIdx.x;
  const int lane = tid & 31, wv = tid >> 5;
  const float* xb = xyz + (size_t)b * cK * 3;
  float px[32], py[32], pz[32], md[32];
#pragma unroll
  for (int i = 0; i < 32; ++i) {
    int kk = tid + i * 256;
    px[i] = xb[kk * 3 + 0]; py[i] = xb[kk * 3 + 1]; pz[i] = xb[kk * 3 + 2];
    md[i] = 1e10f;
  }
  __shared__ float scent[3];
  __shared__ float swv[8];
  __shared__ int   swi[8];
  int far = 0;
  for (int it = 0; it < cP; ++it) {
    if (tid == 0) {
      inds[b * cP + it] = far;
      float cx = xb[far * 3 + 0], cy = xb[far * 3 + 1], cz = xb[far * 3 + 2];
      nxyz[(size_t)(b * cP + it) * 3 + 0] = cx;
      nxyz[(size_t)(b * cP + it) * 3 + 1] = cy;
      nxyz[(size_t)(b * cP + it) * 3 + 2] = cz;
      scent[0] = cx; scent[1] = cy; scent[2] = cz;
    }
    __syncthreads();
    float cx = scent[0], cy = scent[1], cz = scent[2];
    float bv = -1.0f; int bi = 0;
#pragma unroll
    for (int i = 0; i < 32; ++i) {
      float dx = px[i] - cx, dy = py[i] - cy, dz = pz[i] - cz;
      float d = dx * dx + dy * dy + dz * dz;
      md[i] = fminf(md[i], d);
      if (md[i] > bv) { bv = md[i]; bi = tid + i * 256; }
    }
#pragma unroll
    for (int off = 16; off > 0; off >>= 1) {
      float ov = __shfl_down(bv, off);
      int   oi = __shfl_down(bi, off);
      if (ov > bv || (ov == bv && oi < bi)) { bv = ov; bi = oi; }
    }
    if (lane == 0) { swv[wv] = bv; swi[wv] = bi; }
    __syncthreads();
    if (tid == 0) {
      float best = swv[0]; int besti = swi[0];
      for (int w = 1; w < 8; ++w)
        if (swv[w] > best || (swv[w] == best && swi[w] < besti)) { best = swv[w]; besti = swi[w]; }
      far = besti;
    }
  }
}

// ---------------------------------------------------------------------------
// Ball query: one wave32 per query point; ordered first-16 selection.
// ---------------------------------------------------------------------------
__global__ __launch_bounds__(256) void ballq_kernel(const float* __restrict__ xyz,
                                                    const float* __restrict__ nxyz,
                                                    int* __restrict__ idxg) {
  const int gw = (blockIdx.x * blockDim.x + threadIdx.x) >> 5;
  const int lane = threadIdx.x & 31;
  const int b = gw / cP;
  const float* xb = xyz + (size_t)b * cK * 3;
  const float qx = nxyz[(size_t)gw * 3 + 0];
  const float qy = nxyz[(size_t)gw * 3 + 1];
  const float qz = nxyz[(size_t)gw * 3 + 2];
  const float r2 = cRAD * cRAD;
  int cnt = 0, first = 0;
  for (int ch = 0; ch < cK / 32; ++ch) {
    int kk = ch * 32 + lane;
    float dx = xb[kk * 3 + 0] - qx;
    float dy = xb[kk * 3 + 1] - qy;
    float dz = xb[kk * 3 + 2] - qz;
    int pred = (dx * dx + dy * dy + dz * dz) <= r2;
    unsigned m = (unsigned)__ballot(pred);
    if (m) {
      if (cnt == 0) first = ch * 32 + __builtin_ctz(m);
      int rank = __popc(m & ((1u << lane) - 1u));
      if (pred && (cnt + rank) < cS) idxg[(size_t)gw * cS + cnt + rank] = kk;
      cnt += __popc(m);
      if (cnt >= cS) break;
    }
  }
  for (int l = cnt + lane; l < cS; l += 32) idxg[(size_t)gw * cS + l] = first;
}

// ---------------------------------------------------------------------------
// One 16-row x 64-col WMMA layer: A-frags from global bf16 weights,
// B-frags from LDS (sample-major [n][K]), relu + bf16 store to LDS.
// ---------------------------------------------------------------------------
__device__ __forceinline__ void gemm_relu_store(
    const u16* __restrict__ Wg, int KPW, const float* __restrict__ bias,
    const u16* src, int KPs, int ksteps, u16* dst, int KPd) {
  const int tid = threadIdx.x;
  const int wv = tid >> 5, lane = tid & 31, hi = lane >> 4, r = lane & 15;
  v8f z = {0.f, 0.f, 0.f, 0.f, 0.f, 0.f, 0.f, 0.f};
  v8f acc[4];
#pragma unroll
  for (int ct = 0; ct < 4; ++ct) acc[ct] = z;
  for (int ks = 0; ks < ksteps; ++ks) {
    FragU a;
    const u16* wrow = Wg + (size_t)(wv * 16 + r) * KPW + ks * 32;
#pragma unroll
    for (int vv = 0; vv < 8; ++vv) {
      int kb = 2 * vv + 16 * hi;   // contiguous k order, matches B
      a.u[vv] = *(const u32*)(wrow + kb);
    }
#pragma unroll
    for (int ct = 0; ct < 4; ++ct) {
      FragU bb;
      const u16* srow = src + (ct * 16 + r) * KPs + ks * 32 + 16 * hi;
#pragma unroll
      for (int vv = 0; vv < 8; ++vv) bb.u[vv] = *(const u32*)(srow + 2 * vv);
      acc[ct] = __builtin_amdgcn_wmma_f32_16x16x32_bf16(
          false, a.v, false, bb.v, (short)0, acc[ct], false, false);
    }
  }
  const int obase = wv * 16 + 8 * hi;
  float bs[8];
#pragma unroll
  for (int j = 0; j < 8; ++j) bs[j] = bias[obase + j];
#pragma unroll
  for (int ct = 0; ct < 4; ++ct) {
    union { u16 s[8]; uint4 q; } pk;
#pragma unroll
    for (int j = 0; j < 8; ++j) {
      float f = fmaxf(acc[ct][j] + bs[j], 0.0f);
      pk.s[j] = f2bf(f);
    }
    *(uint4*)(dst + (ct * 16 + r) * KPd + obase) = pk.q;
  }
}

// ---------------------------------------------------------------------------
// Gather + 3 fused MLP layers + max-pool over S. WG = 4 points (64 cols).
// ---------------------------------------------------------------------------
__global__ __launch_bounds__(256) void pointmlp_kernel(
    const float* __restrict__ feats, const float* __restrict__ xyz,
    const int* __restrict__ idxg, const float* __restrict__ nxyz,
    const u16* __restrict__ Wb0, const float* __restrict__ Bb0,
    const u16* __restrict__ Wb1, const float* __restrict__ Bb1,
    const u16* __restrict__ Wb2, const float* __restrict__ Bb2,
    u16* __restrict__ featg) {
  __shared__ u16 sX[64 * KP0];   // 20 KB
  __shared__ u16 sY[64 * CH];    // 16 KB
  __shared__ int skid[64];
  const int tid = threadIdx.x;
  const int b = blockIdx.x / (cP / 4);
  const int ptbase = (blockIdx.x % (cP / 4)) * 4;
  if (tid < 64)
    skid[tid] = idxg[(size_t)(b * cP + ptbase + (tid >> 4)) * cS + (tid & 15)];
  __syncthreads();
  for (int pos = tid; pos < 64 * KP0; pos += 256) {
    int n = pos / KP0, c = pos - n * KP0;
    int kk = skid[n];
    float v;
    if (c < 3)
      v = (xyz[((size_t)b * cK + kk) * 3 + c] -
           nxyz[(size_t)(b * cP + ptbase + (n >> 4)) * 3 + c]) * (1.0f / cRAD);
    else if (c < 131)
      v = feats[(size_t)(b * cC + (c - 3)) * cK + kk];
    else
      v = 0.0f;
    sX[pos] = f2bf(v);
  }
  __syncthreads();
  gemm_relu_store(Wb0, KP0, Bb0, sX, KP0, 5, sY, CH);    // layer 0: K=160
  __syncthreads();
  gemm_relu_store(Wb1, CH, Bb1, sY, CH, 4, sX, KP0);     // layer 1: K=128
  __syncthreads();
  // layer 2 + max over S (each col-tile == one point's 16 samples)
  const int wv = tid >> 5, lane = tid & 31, hi = lane >> 4, r = lane & 15;
  v8f z = {0.f, 0.f, 0.f, 0.f, 0.f, 0.f, 0.f, 0.f};
  v8f acc[4];
#pragma unroll
  for (int ct = 0; ct < 4; ++ct) acc[ct] = z;
  for (int ks = 0; ks < 4; ++ks) {
    FragU a;
    const u16* wrow = Wb2 + (size_t)(wv * 16 + r) * CH + ks * 32;
#pragma unroll
    for (int vv = 0; vv < 8; ++vv) {
      int kb = 2 * vv + 16 * hi;
      a.u[vv] = *(const u32*)(wrow + kb);
    }
#pragma unroll
    for (int ct = 0; ct < 4; ++ct) {
      FragU bb;
      const u16* srow = sX + (ct * 16 + r) * KP0 + ks * 32 + 16 * hi;
#pragma unroll
      for (int vv = 0; vv < 8; ++vv) bb.u[vv] = *(const u32*)(srow + 2 * vv);
      acc[ct] = __builtin_amdgcn_wmma_f32_16x16x32_bf16(
          false, a.v, false, bb.v, (short)0, acc[ct], false, false);
    }
  }
  const int obase = wv * 16 + 8 * hi;
  float bs[8];
#pragma unroll
  for (int j = 0; j < 8; ++j) bs[j] = Bb2[obase + j];
#pragma unroll
  for (int ct = 0; ct < 4; ++ct) {
    float f[8];
#pragma unroll
    for (int j = 0; j < 8; ++j) f[j] = fmaxf(acc[ct][j] + bs[j], 0.0f);
#pragma unroll
    for (int j = 0; j < 8; ++j) {
#pragma unroll
      for (int off = 1; off < 16; off <<= 1)          // reduce within 16-lane halves
        f[j] = fmaxf(f[j], __shfl_xor(f[j], off));
    }
    if (r == 0) {
      union { u16 s[8]; uint4 q; } pk;
#pragma unroll
      for (int j = 0; j < 8; ++j) pk.s[j] = f2bf(f[j]);
      *(uint4*)(featg + (size_t)(b * cP + ptbase + ct) * CH + obase) = pk.q;
    }
  }
}

// ---------------------------------------------------------------------------
// Heads: 2 fused 128x128 layers + 80x128 output layer + postprocess.
// WG = 64 points.
// ---------------------------------------------------------------------------
__global__ __launch_bounds__(256) void heads_kernel(
    const u16* __restrict__ featg, const float* __restrict__ nxyz,
    const float* __restrict__ msa,
    const u16* __restrict__ Wh1, const float* __restrict__ Bh1,
    const u16* __restrict__ Wh2, const float* __restrict__ Bh2,
    const u16* __restrict__ Wh3, const float* __restrict__ Bh3,
    float* __restrict__ outp) {
  __shared__ u16 sA[64 * CH];
  __shared__ u16 sB[64 * CH];
  const int tid = threadIdx.x;
  const int b = blockIdx.x / (cP / 64);
  const int ptbase = (blockIdx.x % (cP / 64)) * 64;
  {
    const uint4* srcq = (const uint4*)(featg + (size_t)(b * cP + ptbase) * CH);
    uint4* dstq = (uint4*)sA;
    for (int i = tid; i < 64 * CH / 8; i += 256) dstq[i] = srcq[i];
  }
  __syncthreads();
  gemm_relu_store(Wh1, CH, Bh1, sA, CH, 4, sB, CH);
  __syncthreads();
  gemm_relu_store(Wh2, CH, Bh2, sB, CH, 4, sA, CH);
  __syncthreads();
  const int wv = tid >> 5, lane = tid & 31, hi = lane >> 4, r = lane & 15;
  v8f z = {0.f, 0.f, 0.f, 0.f, 0.f, 0.f, 0.f, 0.f};
  for (int job = wv; job < 20; job += 8) {          // 5 row-tiles x 4 col-tiles
    const int rt = job >> 2, ct = job & 3;
    v8f acc = z;
    for (int ks = 0; ks < 4; ++ks) {
      FragU a;
      const u16* wrow = Wh3 + (size_t)(rt * 16 + r) * CH + ks * 32;
#pragma unroll
      for (int vv = 0; vv < 8; ++vv) {
        int kb = 2 * vv + 16 * hi;
        a.u[vv] = *(const u32*)(wrow + kb);
      }
      FragU bb;
      const u16* srow = sA + (ct * 16 + r) * CH + ks * 32 + 16 * hi;
#pragma unroll
      for (int vv = 0; vv < 8; ++vv) bb.u[vv] = *(const u32*)(srow + 2 * vv);
      acc = __builtin_amdgcn_wmma_f32_16x16x32_bf16(
          false, a.v, false, bb.v, (short)0, acc, false, false);
    }
    const int obase = rt * 16 + 8 * hi;
    const int pt = ptbase + ct * 16 + r;
#pragma unroll
    for (int j = 0; j < 8; ++j) {
      int o = obase + j;
      if (o >= cOUT) continue;
      float val = acc[j] + Bh3[o];
      float res;
      if (o < 2)       res = val;                                            // obj
      else if (o < 5)  res = val + nxyz[(size_t)(b * cP + pt) * 3 + (o - 2)];// center
      else if (o < 17) res = val;                                            // hs
      else if (o < 29) res = val * (3.14159265358979323846f / 12.0f);        // hr
      else if (o < 39) res = val;                                            // ss
      else if (o < 69) res = val * msa[o - 39];                              // sr
      else             res = val;                                            // sem
      outp[(size_t)(b * cP + pt) * cOUT + o] = res;
    }
  }
}

// ---------------------------------------------------------------------------
extern "C" void kernel_launch(void* const* d_in, const int* in_sizes, int n_in,
                              void* d_out, int out_size, void* d_ws, size_t ws_size,
                              hipStream_t stream) {
  (void)in_sizes; (void)n_in; (void)out_size; (void)ws_size;
  const float* xyz  = (const float*)d_in[0];
  const float* feat = (const float*)d_in[1];
  const float* msa  = (const float*)d_in[2];
  const float* w0   = (const float*)d_in[3];
  const float* w1   = (const float*)d_in[4];
  const float* w2   = (const float*)d_in[5];
  const float* g0   = (const float*)d_in[6];
  const float* b0   = (const float*)d_in[7];
  const float* g1   = (const float*)d_in[8];
  const float* b1   = (const float*)d_in[9];
  const float* g2   = (const float*)d_in[10];
  const float* b2   = (const float*)d_in[11];
  const float* wc1  = (const float*)d_in[12];
  const float* bc1  = (const float*)d_in[13];
  const float* gn1  = (const float*)d_in[14];
  const float* bn1  = (const float*)d_in[15];
  const float* wc2  = (const float*)d_in[16];
  const float* bc2  = (const float*)d_in[17];
  const float* gn2  = (const float*)d_in[18];
  const float* bn2  = (const float*)d_in[19];
  const float* wc3  = (const float*)d_in[20];
  const float* bc3  = (const float*)d_in[21];

  char* ws = (char*)d_ws;
  int*   inds  = (int*)(ws + OFF_IND);
  float* nxyz  = (float*)(ws + OFF_NXYZ);
  int*   idxg  = (int*)(ws + OFF_IDX);
  u16*   Wb0   = (u16*)(ws + OFF_WB0);  float* Bb0 = (float*)(ws + OFF_BB0);
  u16*   Wb1   = (u16*)(ws + OFF_WB1);  float* Bb1 = (float*)(ws + OFF_BB1);
  u16*   Wb2   = (u16*)(ws + OFF_WB2);  float* Bb2 = (float*)(ws + OFF_BB2);
  u16*   Wh1   = (u16*)(ws + OFF_WH1);  float* Bh1 = (float*)(ws + OFF_BH1);
  u16*   Wh2   = (u16*)(ws + OFF_WH2);  float* Bh2 = (float*)(ws + OFF_BH2);
  u16*   Wh3   = (u16*)(ws + OFF_WH3);  float* Bh3 = (float*)(ws + OFF_BH3);
  u16*   featg = (u16*)(ws + OFF_FEAT);

  prep_kernel<<<96, 256, 0, stream>>>(w0, w1, w2, g0, b0, g1, b1, g2, b2,
                                      wc1, bc1, gn1, bn1, wc2, bc2, gn2, bn2,
                                      wc3, bc3,
                                      Wb0, Bb0, Wb1, Bb1, Wb2, Bb2,
                                      Wh1, Bh1, Wh2, Bh2, Wh3, Bh3);
  fps_kernel<<<cB, 256, 0, stream>>>(xyz, inds, nxyz);
  ballq_kernel<<<cB * cP / 8, 256, 0, stream>>>(xyz, nxyz, idxg);
  pointmlp_kernel<<<cB * (cP / 4), 256, 0, stream>>>(feat, xyz, idxg, nxyz,
                                                     Wb0, Bb0, Wb1, Bb1, Wb2, Bb2,
                                                     featg);
  heads_kernel<<<cB * (cP / 64), 256, 0, stream>>>(featg, nxyz, msa,
                                                   Wh1, Bh1, Wh2, Bh2, Wh3, Bh3,
                                                   (float*)d_out);
}
